// AGSMLayer_51135880626466
// MI455X (gfx1250) — matrix operand order repaired and checked by reading, not compiled, not run on hardware
//
#include <hip/hip_runtime.h>
#include <hip/hip_bf16.h>
#include <math.h>

// ---------------- problem constants ----------------
#define BATCH 8
#define SEQ   2048
#define DM    256
#define DFF   1024
#define MTOK  (BATCH * SEQ)          // 16384
#define BETA_C  12.0f
#define SCALE_C 0.0625f              // 1/sqrt(256)

// ---------------- CDNA5 async global->LDS support ----------------
#if __has_builtin(__builtin_amdgcn_global_load_async_to_lds_b128) && \
    __has_builtin(__builtin_amdgcn_s_wait_asynccnt)
#define HAVE_ASYNC_LDS 1
#else
#define HAVE_ASYNC_LDS 0
#endif

// ---------------- CDNA5 vector types ----------------
typedef __attribute__((ext_vector_type(16))) __bf16          v16bf;
typedef __attribute__((ext_vector_type(8)))  float           v8f;
typedef __attribute__((ext_vector_type(16))) unsigned short  v16u;
typedef __attribute__((ext_vector_type(8)))  unsigned short  v8us;

#if HAVE_ASYNC_LDS
// builtin signature (from hipcc diagnostic): param0 = AS1 int __vector(4) *
typedef int v4i_vs __attribute__((vector_size(16)));
typedef __attribute__((address_space(1))) v4i_vs as1_v4i;
typedef __attribute__((address_space(3))) v4i_vs as3_v4i;
// generic LDS pointer: addr[31:0] is the LDS offset (ISA 10.2 aperture rules)
static __device__ inline void async_cp16(const void* g, void* l) {
    __builtin_amdgcn_global_load_async_to_lds_b128(
        (as1_v4i*)(unsigned long long)g,
        (as3_v4i*)(unsigned int)(unsigned long long)l, 0, 0);
}
#endif

static __device__ inline v8f zero8() {
    v8f z;
#pragma unroll
    for (int i = 0; i < 8; ++i) z[i] = 0.0f;
    return z;
}

// round-to-nearest-even f32 -> bf16 (as u16 storage)
static __device__ inline unsigned short f2bf(float f) {
    unsigned int u = __float_as_uint(f);
    u += 0x7FFFu + ((u >> 16) & 1u);
    return (unsigned short)(u >> 16);
}

// Build a 16-element bf16 fragment from two contiguous 8xu16 (16B) chunks.
static __device__ inline v16bf load_frag(const unsigned short* p0,
                                         const unsigned short* p1) {
    v8us lo = *reinterpret_cast<const v8us*>(p0);
    v8us hi = *reinterpret_cast<const v8us*>(p1);
    v16u c  = __builtin_shufflevector(lo, hi,
                0,1,2,3,4,5,6,7,8,9,10,11,12,13,14,15);
    return __builtin_bit_cast(v16bf, c);
}

static __device__ inline float gelu_f(float x) {
    return 0.5f * x * (1.0f + erff(x * 0.70710678118f));
}
static __device__ inline float sigmoid_f(float x) {
    return 1.0f / (1.0f + __expf(-x));
}

// ---------------- elementwise prep kernels ----------------
__global__ void cvt_f32_bf16_kernel(const float* __restrict__ in,
                                    unsigned short* __restrict__ out, int n) {
    int i = blockIdx.x * blockDim.x + threadIdx.x;
    if (i < n) out[i] = f2bf(in[i]);
}

__global__ void gpow_kernel(const float* __restrict__ g,
                            float* __restrict__ gp, int n) {
    int i = blockIdx.x * blockDim.x + threadIdx.x;
    if (i < n) gp[i] = powf(g[i], 0.8f);
}

// Vt[b][d][n] = qkv[(b*SEQ+n)*768 + 512 + d]   (bf16 storage)
__global__ void vtrans_kernel(const unsigned short* __restrict__ qkv,
                              unsigned short* __restrict__ vt) {
    int i = blockIdx.x * blockDim.x + threadIdx.x;   // over B*DM*SEQ
    int n = i & (SEQ - 1);
    int t = i >> 11;
    int d = t & (DM - 1);
    int b = t >> 8;
    vt[i] = qkv[((size_t)(b * SEQ + n)) * 768 + 512 + d];
}

// ---------------- LayerNorm (+ optional tau) ----------------
template <bool WITH_TAU>
__global__ __launch_bounds__(256) void ln_kernel(
    const float* __restrict__ x, const float* __restrict__ w,
    const float* __restrict__ b, const float* __restrict__ tau_w,
    const float* __restrict__ tau_b, unsigned short* __restrict__ h_out,
    float* __restrict__ tau_out) {
    __shared__ float red[256];
    const int tok = blockIdx.x;
    const int i   = threadIdx.x;
    float v = x[(size_t)tok * DM + i];
    red[i] = v; __syncthreads();
#pragma unroll
    for (int s = 128; s > 0; s >>= 1) { if (i < s) red[i] += red[i + s]; __syncthreads(); }
    float mean = red[0] * (1.0f / DM); __syncthreads();
    float d = v - mean;
    red[i] = d * d; __syncthreads();
#pragma unroll
    for (int s = 128; s > 0; s >>= 1) { if (i < s) red[i] += red[i + s]; __syncthreads(); }
    float var = red[0] * (1.0f / DM); __syncthreads();
    float h = d * rsqrtf(var + 1e-5f) * w[i] + b[i];
    h_out[(size_t)tok * DM + i] = f2bf(h);
    if (WITH_TAU) {
        red[i] = h * tau_w[i]; __syncthreads();
#pragma unroll
        for (int s = 128; s > 0; s >>= 1) { if (i < s) red[i] += red[i + s]; __syncthreads(); }
        if (i == 0) tau_out[tok] = red[0] + tau_b[0];
    }
}

// ---------------- generic bf16 WMMA GEMM (double-buffered LDS) ----------------
// C[M,N] = A[M,K](bf16) * B[K,N](bf16) [+bias] [+gelu] [+res], out f32 or bf16
#define BM 64
#define BN 64
#define BK 32

template <bool OUT_BF16, bool GELU>
__global__ __launch_bounds__(128) void gemm_bf16_kernel(
    const unsigned short* __restrict__ A, const unsigned short* __restrict__ B,
    const float* __restrict__ bias, const float* __restrict__ res,
    float* __restrict__ outF, unsigned short* __restrict__ outB,
    int M, int N, int K) {
    __shared__ __align__(16) unsigned short As[2][BM][BK];
    __shared__ __align__(16) unsigned short Bs[2][BN][BK];   // transposed tiles
    const int tid  = threadIdx.x;
    const int wave = tid >> 5;
    const int lane = tid & 31;
    const int half = lane >> 4;
    const int l16  = lane & 15;
    const int mBase = blockIdx.y * BM;
    const int nBase = blockIdx.x * BN;

    auto stage = [&](int buf, int kk) {
        // A: 64x32, 32B per thread -> two async/vector 16B copies
        int row = tid >> 1;
        int c0  = (tid & 1) * 16;
        const unsigned short* srcA = A + (size_t)(mBase + row) * K + kk + c0;
#if HAVE_ASYNC_LDS
        async_cp16(srcA,     &As[buf][row][c0]);
        async_cp16(srcA + 8, &As[buf][row][c0 + 8]);
#else
        v8us a0 = *reinterpret_cast<const v8us*>(srcA);
        v8us a1 = *reinterpret_cast<const v8us*>(srcA + 8);
        *reinterpret_cast<v8us*>(&As[buf][row][c0])     = a0;
        *reinterpret_cast<v8us*>(&As[buf][row][c0 + 8]) = a1;
#endif
        // B transposed: Bs[n][k] (scatter, keeps fragments contiguous)
        int krow = tid >> 2;
        int n0   = (tid & 3) * 16;
        const unsigned short* srcB = B + (size_t)(kk + krow) * N + nBase + n0;
#pragma unroll
        for (int i = 0; i < 16; ++i) Bs[buf][n0 + i][krow] = srcB[i];
    };

    v8f acc[4];
#pragma unroll
    for (int t = 0; t < 4; ++t) acc[t] = zero8();

    const int nsteps = K / BK;
    stage(0, 0);
#if HAVE_ASYNC_LDS
    __builtin_amdgcn_s_wait_asynccnt(0);
#endif
    __syncthreads();
    for (int s = 0; s < nsteps; ++s) {
        const int cur = s & 1;
        if (s + 1 < nsteps) stage(cur ^ 1, (s + 1) * BK);   // overlap with compute
        const int r0 = wave * 16;
        v16bf af = load_frag(&As[cur][r0 + l16][half * 16],
                             &As[cur][r0 + l16][half * 16 + 8]);
#pragma unroll
        for (int t = 0; t < 4; ++t) {
            v16bf bf = load_frag(&Bs[cur][t * 16 + l16][half * 16],
                                 &Bs[cur][t * 16 + l16][half * 16 + 8]);
            acc[t] = __builtin_amdgcn_wmma_f32_16x16x32_bf16(
                false, af, false, bf, (short)0, acc[t], false, false);
        }
#if HAVE_ASYNC_LDS
        __builtin_amdgcn_s_wait_asynccnt(0);   // next tile landed (after compute)
#endif
        __syncthreads();
    }
    // epilogue: C layout VGPR j -> row (wave*16 + j + half*8), col (t*16 + l16)
#pragma unroll
    for (int t = 0; t < 4; ++t) {
#pragma unroll
        for (int j = 0; j < 8; ++j) {
            int row = mBase + wave * 16 + half * 8 + j;
            int col = nBase + t * 16 + l16;
            float v = acc[t][j];
            if (bias) v += bias[col];
            if (GELU) v = gelu_f(v);
            if (res)  v += res[(size_t)row * N + col];
            if (OUT_BF16) outB[(size_t)row * N + col] = f2bf(v);
            else          outF[(size_t)row * N + col] = v;
        }
    }
}

// ---------------- fused gated flash attention ----------------
// 1 wave per 16 queries. K/V tiles staged via async global->LDS (one ASYNCcnt
// wait per 32-key chunk instead of per-fragment loadcnt drains), scores via
// bf16 WMMA, gated A written non-temporally, online sigmoid-masked softmax
// (softmax Z cancels in the renormalization), O += P V via WMMA.
__global__ __launch_bounds__(32) void attn_kernel(
    const unsigned short* __restrict__ qkv,   // [MTOK, 768] bf16: q|k|v
    const unsigned short* __restrict__ vt,    // [B, DM, SEQ] bf16
    const float* __restrict__ tau,            // [MTOK]
    const float* __restrict__ gp,             // [MTOK] = g^0.8
    float* __restrict__ Aout,                 // [B, SEQ, SEQ]
    unsigned short* __restrict__ Oout) {      // [MTOK, DM] bf16
    const int b     = blockIdx.x >> 7;
    const int qbase = (blockIdx.x & 127) * 16;
    const int lane  = threadIdx.x;
    const int half  = lane >> 4;
    const int l16   = lane & 15;

    __shared__ __align__(16) unsigned short Plds[16][32];
#if HAVE_ASYNC_LDS
    __shared__ __align__(16) unsigned short Klds[32][256];   // 32 keys x 256 d
    __shared__ __align__(16) unsigned short Vlds[256][32];   // 256 d x 32 keys
#endif

    // preload Q fragments (A-operand): row = qbase + l16, 8 k-chunks of 32
    v16bf qf[8];
    {
        const unsigned short* qrow = qkv + (size_t)(b * SEQ + qbase + l16) * 768;
#pragma unroll
        for (int kc = 0; kc < 8; ++kc) {
            const unsigned short* p = qrow + kc * 32;
            qf[kc] = load_frag(p + half * 16, p + half * 16 + 8);
        }
    }
    // per-row (this half's 8 rows) scalars
    float tauR[8], gqR[8], mrun[8], srun[8];
#pragma unroll
    for (int j = 0; j < 8; ++j) {
        int row = b * SEQ + qbase + half * 8 + j;
        tauR[j] = tau[row];
        gqR[j]  = gp[row];
        mrun[j] = -INFINITY;
        srun[j] = 0.0f;
    }
    v8f Oacc[16];
#pragma unroll
    for (int c = 0; c < 16; ++c) Oacc[c] = zero8();

    const unsigned short* kbase_p = qkv + (size_t)b * SEQ * 768 + 256; // K block
    for (int kb = 0; kb < SEQ; kb += 32) {
        v8f s0 = zero8(), s1 = zero8();
#if HAVE_ASYNC_LDS
        // ---- async-stage K tile: 32 rows x 512B (lane covers bytes lane*16) ----
#pragma unroll
        for (int i = 0; i < 32; ++i) {
            const unsigned short* gk = kbase_p + (size_t)(kb + i) * 768 + lane * 8;
            async_cp16(gk, &Klds[i][lane * 8]);
        }
        // ---- async-stage V tile: Vlds[d][k]; inst i covers d = i*8 .. i*8+7 ----
#pragma unroll
        for (int i = 0; i < 32; ++i) {
            int d = i * 8 + (lane >> 2);
            const unsigned short* gv =
                vt + ((size_t)b * DM + d) * SEQ + kb + (lane & 3) * 8;
            async_cp16(gv, &Vlds[d][(lane & 3) * 8]);
        }
        __builtin_amdgcn_s_wait_asynccnt(0);
        __syncthreads();
        // ---- scores from LDS fragments ----
#pragma unroll
        for (int kc = 0; kc < 8; ++kc) {
            v16bf kf = load_frag(&Klds[l16][kc * 32 + half * 16],
                                 &Klds[l16][kc * 32 + half * 16 + 8]);
            s0 = __builtin_amdgcn_wmma_f32_16x16x32_bf16(
                false, qf[kc], false, kf, (short)0, s0, false, false);
        }
#pragma unroll
        for (int kc = 0; kc < 8; ++kc) {
            v16bf kf = load_frag(&Klds[16 + l16][kc * 32 + half * 16],
                                 &Klds[16 + l16][kc * 32 + half * 16 + 8]);
            s1 = __builtin_amdgcn_wmma_f32_16x16x32_bf16(
                false, qf[kc], false, kf, (short)0, s1, false, false);
        }
#else
        // ---- fallback: direct global register fragments ----
#pragma unroll
        for (int kc = 0; kc < 8; ++kc) {
            const unsigned short* k0 =
                kbase_p + (size_t)(kb + l16) * 768 + kc * 32 + half * 16;
            s0 = __builtin_amdgcn_wmma_f32_16x16x32_bf16(
                false, qf[kc], false, load_frag(k0, k0 + 8), (short)0, s0, false, false);
        }
#pragma unroll
        for (int kc = 0; kc < 8; ++kc) {
            const unsigned short* k1 =
                kbase_p + (size_t)(kb + 16 + l16) * 768 + kc * 32 + half * 16;
            s1 = __builtin_amdgcn_wmma_f32_16x16x32_bf16(
                false, qf[kc], false, load_frag(k1, k1 + 8), (short)0, s1, false, false);
        }
#endif
        // ---- gate + emit A (non-temporal) + online masked softmax ----
        float gk0 = gp[b * SEQ + kb + l16];
        float gk1 = gp[b * SEQ + kb + 16 + l16];
        float alphaV[8];
#pragma unroll
        for (int j = 0; j < 8; ++j) {
            float a0 = s0[j] * SCALE_C * gqR[j] * gk0;
            float a1 = s1[j] * SCALE_C * gqR[j] * gk1;
            size_t arow = (size_t)(b * SEQ + qbase + half * 8 + j) * SEQ;
            __builtin_nontemporal_store(a0, &Aout[arow + kb + l16]);
            __builtin_nontemporal_store(a1, &Aout[arow + kb + 16 + l16]);
            float cm = fmaxf(a0, a1);
#pragma unroll
            for (int m = 8; m > 0; m >>= 1) cm = fmaxf(cm, __shfl_xor(cm, m, 16));
            float mnew  = fmaxf(mrun[j], cm);
            float alpha = __expf(mrun[j] - mnew);
            mrun[j] = mnew;
            float p0 = sigmoid_f(BETA_C * (a0 - tauR[j])) * __expf(a0 - mnew);
            float p1 = sigmoid_f(BETA_C * (a1 - tauR[j])) * __expf(a1 - mnew);
            float ps = p0 + p1;
#pragma unroll
            for (int m = 8; m > 0; m >>= 1) ps += __shfl_xor(ps, m, 16);
            srun[j] = srun[j] * alpha + ps;
            alphaV[j] = alpha;
            Plds[half * 8 + j][l16]      = f2bf(p0);
            Plds[half * 8 + j][16 + l16] = f2bf(p1);
        }
#pragma unroll
        for (int c = 0; c < 16; ++c)
#pragma unroll
            for (int j = 0; j < 8; ++j) Oacc[c][j] *= alphaV[j];
        __syncthreads();
        // ---- O += P @ V ----
        v16bf pf = load_frag(&Plds[l16][half * 16], &Plds[l16][half * 16 + 8]);
#if HAVE_ASYNC_LDS
#pragma unroll
        for (int c = 0; c < 16; ++c) {
            v16bf vf = load_frag(&Vlds[c * 16 + l16][half * 16],
                                 &Vlds[c * 16 + l16][half * 16 + 8]);
            Oacc[c] = __builtin_amdgcn_wmma_f32_16x16x32_bf16(
                false, pf, false, vf, (short)0, Oacc[c], false, false);
        }
#else
#pragma unroll
        for (int grp = 0; grp < 2; ++grp) {
            v16bf vf[8];
#pragma unroll
            for (int c = 0; c < 8; ++c) {
                const unsigned short* vr =
                    vt + ((size_t)b * DM + (grp * 8 + c) * 16 + l16) * SEQ + kb + half * 16;
                vf[c] = load_frag(vr, vr + 8);
            }
#pragma unroll
            for (int c = 0; c < 8; ++c)
                Oacc[grp * 8 + c] = __builtin_amdgcn_wmma_f32_16x16x32_bf16(
                    false, pf, false, vf[c], (short)0, Oacc[grp * 8 + c], false, false);
        }
#endif
        __syncthreads();
    }
    // ---- normalize + store O (bf16) ----
#pragma unroll
    for (int c = 0; c < 16; ++c) {
#pragma unroll
        for (int j = 0; j < 8; ++j) {
            int row = b * SEQ + qbase + half * 8 + j;
            float v = Oacc[c][j] / (srun[j] + 1e-20f);
            Oout[(size_t)row * DM + c * 16 + l16] = f2bf(v);
        }
    }
}

// ---------------- launcher ----------------
extern "C" void kernel_launch(void* const* d_in, const int* in_sizes, int n_in,
                              void* d_out, int out_size, void* d_ws, size_t ws_size,
                              hipStream_t stream) {
    const float* x      = (const float*)d_in[0];
    const float* g      = (const float*)d_in[1];
    const float* ln1_w  = (const float*)d_in[2];
    const float* ln1_b  = (const float*)d_in[3];
    const float* qkv_w  = (const float*)d_in[4];
    const float* qkv_b  = (const float*)d_in[5];
    const float* proj_w = (const float*)d_in[6];
    const float* proj_b = (const float*)d_in[7];
    const float* tau_w  = (const float*)d_in[8];
    const float* tau_b  = (const float*)d_in[9];
    const float* ln2_w  = (const float*)d_in[10];
    const float* ln2_b  = (const float*)d_in[11];
    const float* ffn1_w = (const float*)d_in[12];
    const float* ffn1_b = (const float*)d_in[13];
    const float* ffn2_w = (const float*)d_in[14];
    const float* ffn2_b = (const float*)d_in[15];

    char* ws = (char*)d_ws;
    size_t o = 0;
    unsigned short* h_bf   = (unsigned short*)(ws + o); o += (size_t)MTOK * DM * 2;   // also reused as h2
    unsigned short* qkv_bf = (unsigned short*)(ws + o); o += (size_t)MTOK * 768 * 2;
    unsigned short* vt     = (unsigned short*)(ws + o); o += (size_t)BATCH * DM * SEQ * 2;
    unsigned short* o_bf   = (unsigned short*)(ws + o); o += (size_t)MTOK * DM * 2;
    float*          x2     = (float*)(ws + o);          o += (size_t)MTOK * DM * 4;
    unsigned short* u_bf   = (unsigned short*)(ws + o); o += (size_t)MTOK * DFF * 2;
    float*          tauP   = (float*)(ws + o);          o += (size_t)MTOK * 4;
    float*          gpP    = (float*)(ws + o);          o += (size_t)MTOK * 4;
    unsigned short* w_qkv  = (unsigned short*)(ws + o); o += (size_t)DM * 768 * 2;
    unsigned short* w_proj = (unsigned short*)(ws + o); o += (size_t)DM * DM * 2;
    unsigned short* w_ffn1 = (unsigned short*)(ws + o); o += (size_t)DM * DFF * 2;
    unsigned short* w_ffn2 = (unsigned short*)(ws + o); o += (size_t)DFF * DM * 2;

    float* outF = (float*)d_out;
    float* Aout = outF + (size_t)MTOK * DM;   // second tuple output

    // weight conversion + gate prep
    cvt_f32_bf16_kernel<<<(DM * 768 + 255) / 256, 256, 0, stream>>>(qkv_w, w_qkv, DM * 768);
    cvt_f32_bf16_kernel<<<(DM * DM + 255) / 256, 256, 0, stream>>>(proj_w, w_proj, DM * DM);
    cvt_f32_bf16_kernel<<<(DM * DFF + 255) / 256, 256, 0, stream>>>(ffn1_w, w_ffn1, DM * DFF);
    cvt_f32_bf16_kernel<<<(DFF * DM + 255) / 256, 256, 0, stream>>>(ffn2_w, w_ffn2, DFF * DM);
    gpow_kernel<<<MTOK / 256, 256, 0, stream>>>(g, gpP, MTOK);

    // LN1 + tau
    ln_kernel<true><<<MTOK, 256, 0, stream>>>(x, ln1_w, ln1_b, tau_w, tau_b, h_bf, tauP);

    // QKV GEMM: [16384,256] x [256,768] -> bf16
    gemm_bf16_kernel<true, false><<<dim3(768 / BN, MTOK / BM), 128, 0, stream>>>(
        h_bf, w_qkv, qkv_b, nullptr, nullptr, qkv_bf, MTOK, 768, DM);

    // V transpose for PV WMMA B-operand
    vtrans_kernel<<<(BATCH * DM * SEQ) / 256, 256, 0, stream>>>(qkv_bf, vt);

    // fused gated flash attention (writes A to d_out, O bf16 to ws)
    attn_kernel<<<BATCH * (SEQ / 16), 32, 0, stream>>>(qkv_bf, vt, tauP, gpP, Aout, o_bf);

    // proj GEMM + residual x -> x2 (f32)
    gemm_bf16_kernel<false, false><<<dim3(DM / BN, MTOK / BM), 128, 0, stream>>>(
        o_bf, w_proj, proj_b, x, x2, nullptr, MTOK, DM, DM);

    // LN2 -> h2 (reuse h_bf region)
    ln_kernel<false><<<MTOK, 256, 0, stream>>>(x2, ln2_w, ln2_b, nullptr, nullptr, h_bf, nullptr);

    // FFN1 + gelu -> bf16
    gemm_bf16_kernel<true, true><<<dim3(DFF / BN, MTOK / BM), 128, 0, stream>>>(
        h_bf, w_ffn1, ffn1_b, nullptr, nullptr, u_bf, MTOK, DFF, DM);

    // FFN2 + residual x2 -> d_out
    gemm_bf16_kernel<false, false><<<dim3(DM / BN, MTOK / BM), 128, 0, stream>>>(
        u_bf, w_ffn2, ffn2_b, x2, outF, nullptr, MTOK, DM, DFF);
}
